// Attention_15908558865595
// MI455X (gfx1250) — compile-verified, matrix operand order checked by reading, not run on hardware
//
#include <hip/hip_runtime.h>

typedef __attribute__((ext_vector_type(2)))  float    v2f;
typedef __attribute__((ext_vector_type(4)))  float    v4f;
typedef __attribute__((ext_vector_type(8)))  float    v8f;
typedef __attribute__((ext_vector_type(16))) _Float16 v16h;

// ---------------------------------------------------------------------------
// Kernel 1: one wave32 computes vsum[p] = sum_k value[p*1024 + k] for the 32
// (b,h) pairs using WMMA with an all-ones B matrix, writes ctx[p] = vsum/1024.
//
// With B == ones, D[m,n] = sum_k A[m,k]: the K-slot each element occupies is
// irrelevant, so each lane simply streams contiguous chunks of row m = lane&15
// (lanes m and m+16 cover disjoint K positions).
// ---------------------------------------------------------------------------
__global__ __launch_bounds__(32) void attn_reduce_value_wmma(
    const float* __restrict__ value, float* __restrict__ ctx) {
  const int lane = threadIdx.x;   // 0..31, full wave => EXEC all ones for WMMA
  const int m    = lane & 15;
  const int half = lane >> 4;     // 0: lanes 0-15, 1: lanes 16-31

  v8f acc0 = {};                  // rows (pairs) 0..15
  v8f acc1 = {};                  // rows (pairs) 16..31

#if __has_builtin(__builtin_amdgcn_wmma_f32_16x16x4_f32)
  // Full-f32 path: A = 16x4 f32 (2 VGPRs), B = 4x16 ones.
  v2f bones; bones[0] = 1.0f; bones[1] = 1.0f;
  const float* p0 = value + (size_t)m * 1024 + 2 * half;          // pairs 0..15
  const float* p1 = value + (size_t)(16 + m) * 1024 + 2 * half;   // pairs 16..31
#pragma unroll 4
  for (int k = 0; k < 1024; k += 4) {
    v2f a0 = *(const v2f*)(p0 + k);   // lane covers {k+2h, k+2h+1}
    v2f a1 = *(const v2f*)(p1 + k);
    acc0 = __builtin_amdgcn_wmma_f32_16x16x4_f32(
        false, a0, false, bones, (short)0, acc0, false, false);
    acc1 = __builtin_amdgcn_wmma_f32_16x16x4_f32(
        false, a1, false, bones, (short)0, acc1, false, false);
  }
#else
  // Fallback: codegen-confirmed f16 WMMA. Sum error after /1024 is ~1e-5 abs.
  v16h ones16;
#pragma unroll
  for (int i = 0; i < 16; ++i) ones16[i] = (_Float16)1.0f;
  const float* p0 = value + (size_t)m * 1024 + half * 16;
  const float* p1 = p0 + (size_t)16 * 1024;
#pragma unroll 2
  for (int k = 0; k < 1024; k += 32) {
    v16h a0, a1;
#pragma unroll
    for (int i = 0; i < 16; ++i) {
      a0[i] = (_Float16)p0[k + i];
      a1[i] = (_Float16)p1[k + i];
    }
    acc0 = __builtin_amdgcn_wmma_f32_16x16x32_f16(
        false, a0, false, ones16, (short)0, acc0, false, false);
    acc1 = __builtin_amdgcn_wmma_f32_16x16x32_f16(
        false, a1, false, ones16, (short)0, acc1, false, false);
  }
#endif

  // D layout (16x16 f32): VGPR r, lanes 0-15 -> M=r ; lanes 16-31 -> M=r+8.
  // Every column n holds the same row-sum, so lanes 0 and 16 scatter results.
  const float s = 1.0f / 1024.0f;
  if (lane == 0) {
#pragma unroll
    for (int r = 0; r < 8; ++r) {
      ctx[r]      = acc0[r] * s;   // pairs 0..7
      ctx[16 + r] = acc1[r] * s;   // pairs 16..23
    }
  } else if (lane == 16) {
#pragma unroll
    for (int r = 0; r < 8; ++r) {
      ctx[8 + r]  = acc0[r] * s;   // pairs 8..15
      ctx[24 + r] = acc1[r] * s;   // pairs 24..31
    }
  }
}

// ---------------------------------------------------------------------------
// Kernel 2: broadcast ctx[bh] over the 1M-element (S*F) tile per (b,h).
// 134 MB of stores -> pure HBM-store-bound (~5.8us floor at 23.3 TB/s).
// One non-temporal b128 store per thread; 1024 blocks per bh (256 thr * 4 f32
// * 1024 blk = 1,048,576 floats = S*F), so bh is block-uniform.
// ---------------------------------------------------------------------------
__global__ __launch_bounds__(256) void attn_broadcast_fill(
    const float* __restrict__ ctx, float* __restrict__ out) {
  const int bh  = blockIdx.x >> 10;        // 1024 blocks per (b,h) tile
  const float c = ctx[bh];                 // block-uniform -> scalar load
  const size_t i = ((size_t)blockIdx.x * 256 + threadIdx.x);  // float4 index
  v4f v = {c, c, c, c};
  __builtin_nontemporal_store(v, (v4f*)out + i);
}

extern "C" void kernel_launch(void* const* d_in, const int* in_sizes, int n_in,
                              void* d_out, int out_size, void* d_ws, size_t ws_size,
                              hipStream_t stream) {
  // Inputs: d_in[0]=query [2,16,1024], d_in[1]=key [2,16,1,1024],
  //         d_in[2]=value [2,16,1024,1].
  // Softmax of a constant row is uniform 1/1024, so only `value` matters.
  const float* value = (const float*)d_in[2];
  float*       ctx   = (float*)d_ws;      // 32 floats of scratch
  float*       out   = (float*)d_out;     // 33,554,432 floats

  attn_reduce_value_wmma<<<1, 32, 0, stream>>>(value, ctx);

  const int total_f4 = out_size / 4;      // 8,388,608
  const int blocks   = total_f4 / 256;    // 32,768
  attn_broadcast_fill<<<blocks, 256, 0, stream>>>(ctx, out);
}
